// GraphAttentionLayer_19859928777027
// MI455X (gfx1250) — matrix operand order repaired and checked by reading, not compiled, not run on hardware
//
#include <hip/hip_runtime.h>
#include <hip/hip_bf16.h>

// ---------------------------------------------------------------------------
// Graph attention layer for MI455X (gfx1250, wave32, WMMA).
// bf16 WMMA for all matmuls, fp32 softmax / LayerNorm, flash-style attention.
// ---------------------------------------------------------------------------

typedef __attribute__((ext_vector_type(16))) __bf16 v16bf;
typedef __attribute__((ext_vector_type(8)))  __bf16 v8bf;
typedef __attribute__((ext_vector_type(8)))  float  v8f;

#define IN_DIM   256
#define OUT_DIM  256
#define NHEADS   8
#define HDIM     32
#define BATCH    8
#define NNODES   1024
#define ROWS     (BATCH * NNODES)   // 8192
#define LN_EPS   1e-5f
#define QSCALE   0.17677669529663687f  // 1/sqrt(32)

static __device__ __forceinline__ __bf16 f2bf(float f) {
    unsigned u = __builtin_bit_cast(unsigned, f);
    unsigned r = u + 0x7FFFu + ((u >> 16) & 1u);
    unsigned short h = (unsigned short)(r >> 16);
    return __builtin_bit_cast(__bf16, h);
}

static __device__ __forceinline__ v8f wmma_bf16(v16bf a, v16bf b, v8f c) {
    // D = A(16x32 bf16) * B(32x16 bf16) + C(16x16 f32)
    return __builtin_amdgcn_wmma_f32_16x16x32_bf16(
        /*neg_a=*/false, a, /*neg_b=*/false, b,
        /*c_mod=*/(short)0, c, /*reuse_a=*/false, /*reuse_b=*/false);
}

// A-operand (16x32, bf16, row-major source). Lane (g = lane/16, m = lane%16)
// holds row m, K = { g*16 .. g*16+15 } (contiguous, matches B loads).
static __device__ __forceinline__ v16bf load_a(const __bf16* rowbase, int g) {
    v8bf lo = *(const v8bf*)(rowbase + g * 16);
    v8bf hi = *(const v8bf*)(rowbase + g * 16 + 8);
    v16bf a;
#pragma unroll
    for (int i = 0; i < 8; ++i) { a[i] = lo[i]; a[8 + i] = hi[i]; }
    return a;
}

// ---------------------------------------------------------------------------
// Stage 0a: node features fp32 -> bf16
// ---------------------------------------------------------------------------
__global__ void k_cvt_x(const float* __restrict__ X, __bf16* __restrict__ Xb) {
    size_t i = (size_t)blockIdx.x * blockDim.x + threadIdx.x;
    Xb[i] = f2bf(X[i]);
}

// ---------------------------------------------------------------------------
// Stage 0b: weights -> bf16, transposed: Wt[mat][n][k] = W[mat][k][n]
// mat: 0=Wq 1=Wk 2=Wv 3=Wo
// ---------------------------------------------------------------------------
__global__ void k_cvt_w(const float* __restrict__ Wq, const float* __restrict__ Wk,
                        const float* __restrict__ Wv, const float* __restrict__ Wo,
                        __bf16* __restrict__ Wt) {
    int mat  = blockIdx.x >> 8;
    int ncol = blockIdx.x & 255;
    int k    = threadIdx.x;
    const float* W = (mat == 0) ? Wq : (mat == 1 ? Wk : (mat == 2 ? Wv : Wo));
    Wt[((size_t)mat * 256 + ncol) * 256 + k] = f2bf(W[(size_t)k * 256 + ncol]);
}

// ---------------------------------------------------------------------------
// Stage 1: QKV projections.  One 16x16 output tile per wave, K-loop of 8 WMMAs.
// Q is pre-scaled by 1/sqrt(D).  V is written transposed per head:
//   Vt[b][h][d][node]  so attn*V B-operands are contiguous loads.
// ---------------------------------------------------------------------------
__global__ void k_qkv(const __bf16* __restrict__ Xb, const __bf16* __restrict__ Wt,
                      const float* __restrict__ bq, const float* __restrict__ bk,
                      const float* __restrict__ bv,
                      __bf16* __restrict__ Qb, __bf16* __restrict__ Kb,
                      __bf16* __restrict__ Vt) {
    int wid  = threadIdx.x >> 5;
    int lane = threadIdx.x & 31;
    int g    = lane >> 4;
    int mn   = lane & 15;

    int tile = blockIdx.x * 8 + wid;          // 0 .. 24575
    int sel  = tile / (512 * 16);             // 0=Q 1=K 2=V
    int rem  = tile - sel * 512 * 16;
    int mt   = rem >> 4;                      // row tile (16 rows)
    int nt   = rem & 15;                      // col tile (16 cols)

    const __bf16* W = Wt + (size_t)sel * OUT_DIM * IN_DIM;  // [n][k]
    v8f acc = {};
#pragma unroll
    for (int kk = 0; kk < IN_DIM / 32; ++kk) {
        int kbase = kk * 32;
        v16bf a = load_a(Xb + (size_t)(mt * 16 + mn) * IN_DIM + kbase, g);
        v16bf b = *(const v16bf*)(W + (size_t)(nt * 16 + mn) * IN_DIM + kbase + g * 16);
        acc = wmma_bf16(a, b, acc);
    }
    int col   = nt * 16 + mn;
    float bias = (sel == 0) ? bq[col] : (sel == 1 ? bk[col] : bv[col]);

    if (sel == 2) {
#pragma unroll
        for (int r = 0; r < 8; ++r) {
            int row  = mt * 16 + g * 8 + r;
            int bidx = row >> 10, node = row & 1023;
            int h = col >> 5, d = col & 31;
            Vt[((size_t)((bidx * NHEADS + h) * HDIM + d)) * NNODES + node] =
                f2bf(acc[r] + bias);
        }
    } else {
        __bf16* dst = (sel == 0) ? Qb : Kb;
        float sc = (sel == 0) ? QSCALE : 1.0f;
#pragma unroll
        for (int r = 0; r < 8; ++r) {
            int row = mt * 16 + g * 8 + r;
            dst[(size_t)row * OUT_DIM + col] = f2bf((acc[r] + bias) * sc);
        }
    }
}

// ---------------------------------------------------------------------------
// Stage 2: flash attention with edge bias.
// Workgroup = (b, h, 16-query-row tile); 4 waves split the 1024 key columns.
// Per wave, per 32-column pair: 2 score WMMAs, online softmax, 2 P*V WMMAs.
// Cross-wave combine via LDS at the end.  Writes bf16 AO[b][n][h*32+d].
// ---------------------------------------------------------------------------
__global__ void k_attn(const __bf16* __restrict__ Qb, const __bf16* __restrict__ Kb,
                       const __bf16* __restrict__ Vt, const float* __restrict__ EW,
                       const float* __restrict__ We, const float* __restrict__ be,
                       __bf16* __restrict__ AO) {
    __shared__ float Pst[4][16][32];   // per-wave P staging (C-layout -> A-layout)
    __shared__ float accL[4][16][32];  // per-wave unnormalized output
    __shared__ float rmW[4][16];       // per-wave row max
    __shared__ float rsW[4][16];       // per-wave row sum-exp

    int w = threadIdx.x >> 5, lane = threadIdx.x & 31;
    int g = lane >> 4, n = lane & 15;

    int blk = blockIdx.x;              // b*512 + h*64 + qt
    int qt  = blk & 63;
    int h   = (blk >> 6) & 7;
    int b   = blk >> 9;
    int qbase = qt * 16;

    // Q A-operand: fixed for the whole kernel (K = HDIM = 32 -> one operand)
    v16bf Aq = load_a(
        Qb + ((size_t)(b * NNODES + qbase + n)) * OUT_DIM + h * HDIM, g);

    const float we = We[h], bbe = be[h];
    const float* ewrow = EW + ((size_t)b * NNODES + qbase) * NNODES;
    const __bf16* Vhead = Vt + ((size_t)(b * NHEADS + h)) * HDIM * NNODES;

    float rm[8], rs[8];
#pragma unroll
    for (int r = 0; r < 8; ++r) { rm[r] = -3.0e38f; rs[r] = 0.0f; }
    v8f acc0 = {}, acc1 = {};

    for (int pair = 0; pair < 8; ++pair) {
        int c0 = w * 256 + pair * 32;
        // --- score tiles: S = Q K^T (Q pre-scaled by 1/sqrt(D)) ---
        v16bf Bk0 = *(const v16bf*)(
            Kb + ((size_t)(b * NNODES + c0 + n)) * OUT_DIM + h * HDIM + g * 16);
        v16bf Bk1 = *(const v16bf*)(
            Kb + ((size_t)(b * NNODES + c0 + 16 + n)) * OUT_DIM + h * HDIM + g * 16);
        v8f z = {};
        v8f s0 = wmma_bf16(Aq, Bk0, z);
        v8f s1 = wmma_bf16(Aq, Bk1, z);
        // --- edge bias: We[h]*ew[b,row,col] + be[h] ---
#pragma unroll
        for (int r = 0; r < 8; ++r) {
            const float* er = ewrow + (size_t)(g * 8 + r) * NNODES;
            s0[r] += er[c0 + n] * we + bbe;
            s1[r] += er[c0 + 16 + n] * we + bbe;
        }
        // --- online softmax update (rows are shared by the 16 lanes of a group) ---
        float p0[8], p1[8];
#pragma unroll
        for (int r = 0; r < 8; ++r) {
            float t = fmaxf(s0[r], s1[r]);
            t = fmaxf(t, __shfl_xor(t, 1, 32));
            t = fmaxf(t, __shfl_xor(t, 2, 32));
            t = fmaxf(t, __shfl_xor(t, 4, 32));
            t = fmaxf(t, __shfl_xor(t, 8, 32));
            float mnew = fmaxf(rm[r], t);
            float scl  = __expf(rm[r] - mnew);
            float e0   = __expf(s0[r] - mnew);
            float e1   = __expf(s1[r] - mnew);
            float ts   = e0 + e1;
            ts += __shfl_xor(ts, 1, 32);
            ts += __shfl_xor(ts, 2, 32);
            ts += __shfl_xor(ts, 4, 32);
            ts += __shfl_xor(ts, 8, 32);
            rs[r] = rs[r] * scl + ts;
            rm[r] = mnew;
            acc0[r] *= scl;
            acc1[r] *= scl;
            p0[r] = e0;
            p1[r] = e1;
        }
        // --- stage P through LDS: C-layout -> A-layout (intra-wave) ---
#pragma unroll
        for (int r = 0; r < 8; ++r) {
            Pst[w][g * 8 + r][n]      = p0[r];
            Pst[w][g * 8 + r][16 + n] = p1[r];
        }
        // intra-wave ordering of the LDS stores before the gathers below
        __asm__ __volatile__("s_wait_dscnt 0" ::: "memory");
        v16bf Ap;
#pragma unroll
        for (int i = 0; i < 8; ++i) {
            Ap[i]     = f2bf(Pst[w][n][g * 16 + i]);
            Ap[8 + i] = f2bf(Pst[w][n][g * 16 + 8 + i]);
        }
        // --- O += P * V (two 16-col halves of the 32-wide head dim) ---
        int kb = c0 + g * 16;
        v16bf Bv0 = *(const v16bf*)(Vhead + (size_t)(n) * NNODES + kb);
        v16bf Bv1 = *(const v16bf*)(Vhead + (size_t)(16 + n) * NNODES + kb);
        acc0 = wmma_bf16(Ap, Bv0, acc0);
        acc1 = wmma_bf16(Ap, Bv1, acc1);
    }

    // --- publish per-wave partials ---
#pragma unroll
    for (int r = 0; r < 8; ++r) {
        accL[w][g * 8 + r][n]      = acc0[r];
        accL[w][g * 8 + r][16 + n] = acc1[r];
    }
    if (n == 0) {
#pragma unroll
        for (int r = 0; r < 8; ++r) {
            rmW[w][g * 8 + r] = rm[r];
            rsW[w][g * 8 + r] = rs[r];
        }
    }
    __syncthreads();

    // --- cross-wave flash combine: 128 threads cover 16 rows x 32 cols ---
    int row   = threadIdx.x >> 3;
    int dbase = (threadIdx.x & 7) * 4;
    float M = fmaxf(fmaxf(rmW[0][row], rmW[1][row]), fmaxf(rmW[2][row], rmW[3][row]));
    float e0 = __expf(rmW[0][row] - M), e1 = __expf(rmW[1][row] - M);
    float e2 = __expf(rmW[2][row] - M), e3 = __expf(rmW[3][row] - M);
    float S = rsW[0][row] * e0 + rsW[1][row] * e1 + rsW[2][row] * e2 + rsW[3][row] * e3;
    float inv = 1.0f / S;
#pragma unroll
    for (int j = 0; j < 4; ++j) {
        int d = dbase + j;
        float o = accL[0][row][d] * e0 + accL[1][row][d] * e1 +
                  accL[2][row][d] * e2 + accL[3][row][d] * e3;
        AO[((size_t)(b * NNODES + qbase + row)) * OUT_DIM + h * HDIM + d] =
            f2bf(o * inv);
    }
}

// ---------------------------------------------------------------------------
// Stage 3: output projection + bias + residual + LayerNorm (fused).
// Workgroup = 16 rows; 4 waves x 4 N-tiles cover the full 256 columns.
// ---------------------------------------------------------------------------
__global__ void k_oproj_ln(const __bf16* __restrict__ AO, const __bf16* __restrict__ WtO,
                           const float* __restrict__ bo, const float* __restrict__ X,
                           const float* __restrict__ gamma, const float* __restrict__ beta,
                           float* __restrict__ out) {
    __shared__ float xs[16][256];
    __shared__ float red[16][8];
    __shared__ float red2[16][8];

    int w = threadIdx.x >> 5, lane = threadIdx.x & 31;
    int g = lane >> 4, n = lane & 15;
    int mt = blockIdx.x;  // 0..511

    v8f acc[4] = {};
    for (int kk = 0; kk < 8; ++kk) {
        int kbase = kk * 32;
        v16bf a = load_a(AO + (size_t)(mt * 16 + n) * OUT_DIM + kbase, g);
#pragma unroll
        for (int t = 0; t < 4; ++t) {
            int nt = w * 4 + t;
            v16bf bm = *(const v16bf*)(
                WtO + (size_t)(nt * 16 + n) * OUT_DIM + kbase + g * 16);
            acc[t] = wmma_bf16(a, bm, acc[t]);
        }
    }
#pragma unroll
    for (int t = 0; t < 4; ++t) {
        int col  = (w * 4 + t) * 16 + n;
        float bias = bo[col];
#pragma unroll
        for (int r = 0; r < 8; ++r) {
            int row = g * 8 + r;
            float resid = X[(size_t)(mt * 16 + row) * IN_DIM + col];
            xs[row][col] = acc[t][r] + bias + resid;
        }
    }
    __syncthreads();

    // LayerNorm: thread t -> row t/8, 32-element segment t%8
    int row = threadIdx.x >> 3;
    int seg = threadIdx.x & 7;
    float s = 0.0f, s2 = 0.0f;
#pragma unroll
    for (int i = 0; i < 32; ++i) {
        float v = xs[row][seg * 32 + i];
        s += v;
        s2 += v * v;
    }
    red[row][seg] = s;
    red2[row][seg] = s2;
    __syncthreads();
    float ts = 0.0f, ts2 = 0.0f;
#pragma unroll
    for (int i = 0; i < 8; ++i) { ts += red[row][i]; ts2 += red2[row][i]; }
    float mu   = ts * (1.0f / 256.0f);
    float var  = ts2 * (1.0f / 256.0f) - mu * mu;
    float rstd = __frsqrt_rn(var + LN_EPS);
#pragma unroll
    for (int i = 0; i < 32; ++i) {
        int c = seg * 32 + i;
        float v = (xs[row][c] - mu) * rstd;
        out[(size_t)(mt * 16 + row) * OUT_DIM + c] = v * gamma[c] + beta[c];
    }
}

// ---------------------------------------------------------------------------
// Host launcher
// ---------------------------------------------------------------------------
extern "C" void kernel_launch(void* const* d_in, const int* in_sizes, int n_in,
                              void* d_out, int out_size, void* d_ws, size_t ws_size,
                              hipStream_t stream) {
    (void)in_sizes; (void)n_in; (void)out_size; (void)ws_size;
    const float* X     = (const float*)d_in[0];
    const float* EW    = (const float*)d_in[1];
    const float* Wq    = (const float*)d_in[2];
    const float* bq    = (const float*)d_in[3];
    const float* Wk    = (const float*)d_in[4];
    const float* bk    = (const float*)d_in[5];
    const float* Wv    = (const float*)d_in[6];
    const float* bv    = (const float*)d_in[7];
    const float* We    = (const float*)d_in[8];
    const float* be    = (const float*)d_in[9];
    const float* Wo    = (const float*)d_in[10];
    const float* bo    = (const float*)d_in[11];
    const float* gamma = (const float*)d_in[12];
    const float* beta  = (const float*)d_in[13];

    char* ws = (char*)d_ws;
    __bf16* Xb = (__bf16*)(ws);                  // 8192x256 bf16   (4 MB)
    __bf16* Wt = (__bf16*)(ws + 4194304);        // 4x256x256 bf16  (0.5 MB)
    __bf16* Qb = (__bf16*)(ws + 4718592);        // 8192x256 bf16   (4 MB)
    __bf16* Kb = (__bf16*)(ws + 8912896);        // 8192x256 bf16   (4 MB)
    __bf16* Vt = (__bf16*)(ws + 13107200);       // 8x8x32x1024 bf16(4 MB)
    __bf16* AO = (__bf16*)(ws + 17301504);       // 8192x256 bf16   (4 MB)
    float* out = (float*)d_out;

    k_cvt_x<<<ROWS * IN_DIM / 256, 256, 0, stream>>>(X, Xb);
    k_cvt_w<<<4 * 256, 256, 0, stream>>>(Wq, Wk, Wv, Wo, Wt);
    k_qkv<<<3072, 256, 0, stream>>>(Xb, Wt, bq, bk, bv, Qb, Kb, Vt);
    k_attn<<<BATCH * NHEADS * 64, 128, 0, stream>>>(Qb, Kb, Vt, EW, We, be, AO);
    k_oproj_ln<<<ROWS / 16, 128, 0, stream>>>(AO, Wt + 3 * 65536, bo, X, gamma, beta, out);
}
